// ArrowTransformer_42700564857443
// MI455X (gfx1250) — matrix operand order repaired and checked by reading, not compiled, not run on hardware
//
#include <hip/hip_runtime.h>
#include <hip/hip_bf16.h>

// Problem constants (match reference)
#define B_  2
#define S_  1024
#define D_  1024
#define H_  16
#define HD_ 64
#define L_  6
#define V_  256

typedef float  v8f   __attribute__((ext_vector_type(8)));
typedef __bf16 v16bf __attribute__((ext_vector_type(16)));
typedef int    v4i_  __attribute__((ext_vector_type(4)));

#ifndef __has_builtin
#define __has_builtin(x) 0
#endif

#if __has_builtin(__builtin_amdgcn_global_load_async_to_lds_b128)
#define HAVE_ASYNC 1
#else
#define HAVE_ASYNC 0
#endif

__device__ __forceinline__ unsigned short f2bf(float f) {
    unsigned u = __float_as_uint(f);
    u += 0x7FFFu + ((u >> 16) & 1u);       // round-to-nearest-even
    return (unsigned short)(u >> 16);
}

// 16-byte global -> LDS copy (async on CDNA5, sync fallback elsewhere)
__device__ __forceinline__ void ld16_to_lds(const unsigned short* g, unsigned short* l) {
#if HAVE_ASYNC
    v4i_* gp = (v4i_*)g;                   // C-style cast drops const
    v4i_* lp = (v4i_*)l;
    __builtin_amdgcn_global_load_async_to_lds_b128(
        (__attribute__((address_space(1))) v4i_*)gp,
        (__attribute__((address_space(3))) v4i_*)lp, 0, 0);
#else
    *(uint4*)l = *(const uint4*)g;
#endif
}

__device__ __forceinline__ void wait_async_le4() {
#if HAVE_ASYNC
#if __has_builtin(__builtin_amdgcn_s_wait_asynccnt)
    __builtin_amdgcn_s_wait_asynccnt(4);
#else
    asm volatile("s_wait_asynccnt 0x4" ::: "memory");
#endif
#endif
}
__device__ __forceinline__ void wait_async_le0() {
#if HAVE_ASYNC
#if __has_builtin(__builtin_amdgcn_s_wait_asynccnt)
    __builtin_amdgcn_s_wait_asynccnt(0);
#else
    asm volatile("s_wait_asynccnt 0x0" ::: "memory");
#endif
#endif
}
__device__ __forceinline__ void wait_ds0() {
    asm volatile("s_wait_dscnt 0x0" ::: "memory");
}

// ---------------------------------------------------------------------------
// Generic strided-batched bf16 GEMM:  C = A(MxK) * B^T  (+bias) (relu?)
//   A: bf16 [M x K] row-major (lda), B: bf16 [N x K] row-major (ldb)
//   C: f32 or bf16 (outbf); transC: store C transposed ([N x M], ldc = M-dim)
//   Batch z: zo = z / binner, zi = z % binner with two strides per operand.
//   triskip: skip blocks strictly above the diagonal (masked later anyway).
// Templated block tile: BMt x BNt, per-wave tile WMt x WNt, BK=32.
// Threads = (BMt/WMt)*(BNt/WNt)*32; each thread stages 2 A + 2 B 16B chunks.
// Double-buffered LDS, async global->LDS staging, WMMA bf16 16x16x32 core.
// ---------------------------------------------------------------------------
#define BK 32

template <int BMt, int BNt, int WMt, int WNt>
__global__ __launch_bounds__((BMt / WMt) * (BNt / WNt) * 32)
void gemm_bf16_wmma(
    const unsigned short* __restrict__ A, long lda, long sA1, long sA2,
    const unsigned short* __restrict__ Bm, long ldb, long sB1, long sB2,
    void* __restrict__ Cv, long ldc, long sC1, long sC2,
    const float* __restrict__ bias,
    int M, int N, int K, int binner,
    int relu, int outbf, int transC, int triskip)
{
    constexpr int NWN  = BNt / WNt;              // waves along N
    constexpr int NTHR = (BMt / WMt) * NWN * 32; // threads per block
    constexpr int MF   = WMt / 16;               // A frags per wave
    constexpr int NF   = WNt / 16;               // B frags per wave
    constexpr int R2   = NTHR / 4;               // staging row offset (2nd chunk)

    const int m0 = blockIdx.y * BMt;
    const int n0 = blockIdx.x * BNt;
    if (triskip && n0 > m0 + (BMt - 1)) return;  // fully masked tile

    const int z  = blockIdx.z;
    const int zi = z % binner;
    const int zo = z / binner;
    A  += (long)zo * sA2 + (long)zi * sA1;
    Bm += (long)zo * sB2 + (long)zi * sB1;
    const long coff = (long)zo * sC2 + (long)zi * sC1;

    __shared__ __align__(16) unsigned short As[2][BMt][BK];
    __shared__ __align__(16) unsigned short Bs[2][BNt][BK];

    const int tid  = threadIdx.x;
    const int lane = tid & 31;
    const int wave = tid >> 5;
    const int wm   = (wave / NWN) * WMt;
    const int wn   = (wave % NWN) * WNt;

    union FragA   { uint4 u[2]; v16bf v; };
    union FragAcc { v8f v; float f[8]; };
    FragAcc acc[MF][NF];
    #pragma unroll
    for (int i = 0; i < MF; ++i)
        #pragma unroll
        for (int j = 0; j < NF; ++j)
            #pragma unroll
            for (int e = 0; e < 8; ++e) acc[i][j].f[e] = 0.f;

    // ---- per-thread staging addressing: 16B chunks, rows r and r+R2 ----
    const int r   = tid >> 2;             // 0..NTHR/4-1
    const int col = (tid & 3) << 3;       // 0,8,16,24
    const unsigned short* ga0 = A  + (long)(m0 + r) * lda + col;
    const unsigned short* ga1 = ga0 + (long)R2 * lda;
    const unsigned short* gb0 = Bm + (long)(n0 + r) * ldb + col;
    const unsigned short* gb1 = gb0 + (long)R2 * ldb;
    unsigned short* la0[2] = { &As[0][r][col],      &As[1][r][col] };
    unsigned short* la1[2] = { &As[0][r + R2][col], &As[1][r + R2][col] };
    unsigned short* lb0[2] = { &Bs[0][r][col],      &Bs[1][r][col] };
    unsigned short* lb1[2] = { &Bs[0][r + R2][col], &Bs[1][r + R2][col] };

    const int frow  = lane & 15;
    const int akoff = (lane >> 4) * 8;    // A: K halves 0-7/16-23 vs 8-15/24-31
    const int bkoff = (lane >> 4) * 8;    // B: K halves 0-7/16-23 vs 8-15/24-31

    const int nst = K / BK;
    // prologue: stage 0 into buffer 0
    ld16_to_lds(ga0, la0[0]); ld16_to_lds(ga1, la1[0]);
    ld16_to_lds(gb0, lb0[0]); ld16_to_lds(gb1, lb1[0]);

    for (int s = 0; s < nst; ++s) {
        const int buf = s & 1;
        if (s + 1 < nst) {
            const long k1 = (long)(s + 1) * BK;
            ld16_to_lds(ga0 + k1, la0[buf ^ 1]); ld16_to_lds(ga1 + k1, la1[buf ^ 1]);
            ld16_to_lds(gb0 + k1, lb0[buf ^ 1]); ld16_to_lds(gb1 + k1, lb1[buf ^ 1]);
            wait_async_le4();             // stage s has landed (4 newer outstanding)
        } else {
            wait_async_le0();
        }
        __syncthreads();                  // stage s visible to all waves

        FragA a[MF], b[NF];
        #pragma unroll
        for (int t = 0; t < MF; ++t) {
            const int mr = wm + t * 16 + frow;
            a[t].u[0] = *(const uint4*)&As[buf][mr][akoff];
            a[t].u[1] = *(const uint4*)&As[buf][mr][16 + akoff];
        }
        #pragma unroll
        for (int t = 0; t < NF; ++t) {
            const int nr = wn + t * 16 + frow;
            b[t].u[0] = *(const uint4*)&Bs[buf][nr][bkoff];
            b[t].u[1] = *(const uint4*)&Bs[buf][nr][16 + bkoff];
        }
        wait_ds0();                       // my LDS reads done
        __syncthreads();                  // everyone's reads done -> buf reusable

        #pragma unroll
        for (int i = 0; i < MF; ++i)
            #pragma unroll
            for (int j = 0; j < NF; ++j)
                acc[i][j].v = __builtin_amdgcn_wmma_f32_16x16x32_bf16(
                    false, a[i].v, false, b[j].v, (short)0, acc[i][j].v,
                    false, false);
    }

    // ---- epilogue ----
    #pragma unroll
    for (int i = 0; i < MF; ++i) {
        #pragma unroll
        for (int j = 0; j < NF; ++j) {
            const int gn  = n0 + wn + j * 16 + (lane & 15);
            const int gm0 = m0 + wm + i * 16 + ((lane >> 4) << 3);
            const float bv = bias ? bias[gn] : 0.f;
            if (transC) {
                union { unsigned short us[8]; uint4 u; } pk;
                #pragma unroll
                for (int e = 0; e < 8; ++e) {
                    float vv = acc[i][j].f[e] + bv;
                    if (relu) vv = fmaxf(vv, 0.f);
                    pk.us[e] = f2bf(vv);
                }
                *(uint4*)((unsigned short*)Cv + coff + (long)gn * ldc + gm0) = pk.u;
            } else if (outbf) {
                unsigned short* C = (unsigned short*)Cv + coff;
                #pragma unroll
                for (int e = 0; e < 8; ++e) {
                    float vv = acc[i][j].f[e] + bv;
                    if (relu) vv = fmaxf(vv, 0.f);
                    C[(long)(gm0 + e) * ldc + gn] = f2bf(vv);
                }
            } else {
                float* C = (float*)Cv + coff;
                #pragma unroll
                for (int e = 0; e < 8; ++e) {
                    float vv = acc[i][j].f[e] + bv;
                    if (relu) vv = fmaxf(vv, 0.f);
                    C[(long)(gm0 + e) * ldc + gn] = vv;
                }
            }
        }
    }
}

// ---------------------------------------------------------------------------
// Block reductions (256 threads)
// ---------------------------------------------------------------------------
__device__ __forceinline__ float blockReduceMax(float v, float* red) {
    const int tid = threadIdx.x;
    red[tid] = v; __syncthreads();
    for (int s = 128; s > 0; s >>= 1) {
        if (tid < s) red[tid] = fmaxf(red[tid], red[tid + s]);
        __syncthreads();
    }
    float r = red[0]; __syncthreads();
    return r;
}
__device__ __forceinline__ float blockReduceSum(float v, float* red) {
    const int tid = threadIdx.x;
    red[tid] = v; __syncthreads();
    for (int s = 128; s > 0; s >>= 1) {
        if (tid < s) red[tid] += red[tid + s];
        __syncthreads();
    }
    float r = red[0]; __syncthreads();
    return r;
}

// ---------------------------------------------------------------------------
// Transpose + f32->bf16:  in [K x N] f32  ->  out [N x K] bf16  (z = matrix id)
// ---------------------------------------------------------------------------
__global__ __launch_bounds__(256) void transpose_cvt_k(
    const float* __restrict__ in, unsigned short* __restrict__ out,
    int K, int N)
{
    __shared__ float t[32][33];
    const long zo = (long)blockIdx.z * K * N;
    in  += zo; out += zo;
    const int tx = threadIdx.x & 31, ty = threadIdx.x >> 5;   // ty 0..7
    const int k0 = blockIdx.y << 5, n0 = blockIdx.x << 5;
    #pragma unroll
    for (int p = 0; p < 4; ++p)
        t[ty + p * 8][tx] = in[(long)(k0 + ty + p * 8) * N + (n0 + tx)];
    __syncthreads();
    #pragma unroll
    for (int p = 0; p < 4; ++p)
        out[(long)(n0 + ty + p * 8) * K + (k0 + tx)] = f2bf(t[tx][ty + p * 8]);
}

// ---------------------------------------------------------------------------
// erev[l][d][kk] = bf16( E[l][S-1-d][kk] )
// ---------------------------------------------------------------------------
__global__ __launch_bounds__(256) void erev_k(
    const float* __restrict__ E, unsigned short* __restrict__ out)
{
    const int gid = blockIdx.x * 256 + threadIdx.x;   // L*S*HD total
    const int l = gid / (S_ * HD_);
    const int r = gid % (S_ * HD_);
    const int d = r / HD_;
    const int kk = r % HD_;
    out[gid] = f2bf(E[(long)l * S_ * HD_ + (long)(S_ - 1 - d) * HD_ + kk]);
}

// ---------------------------------------------------------------------------
// Embedding + sinusoidal PE -> bf16 h
// ---------------------------------------------------------------------------
__global__ __launch_bounds__(256) void embed_k(
    const int* __restrict__ x, const float* __restrict__ emb,
    unsigned short* __restrict__ h)
{
    const int bs  = blockIdx.x;
    const int tok = x[bs];
    const float pos = (float)(bs & (S_ - 1));
    #pragma unroll
    for (int t = 0; t < 4; ++t) {
        const int d = threadIdx.x + (t << 8);
        const float par  = (float)(d & 1);
        const float rate = __expf(9.210340371976184f * (par - (float)d) *
                                  (1.0f / (float)D_));
        const float pe = __sinf(pos * rate + 1.5707963267948966f * par);
        h[(long)bs * D_ + d] = f2bf(emb[(long)tok * D_ + d] * 32.0f + pe);
    }
}

// ---------------------------------------------------------------------------
// Fused rel-gather + mask + softmax.  Reads qk,qe (f32), writes attn bf16.
// logits[i,j] = (qk[i,j] + (j<=i ? qe[i,i-j] : 0))*scale + mask
// ---------------------------------------------------------------------------
__global__ __launch_bounds__(256) void softmax_rel_k(
    const float* __restrict__ qk, const float* __restrict__ qe,
    const int* __restrict__ x, unsigned short* __restrict__ attn, float scale)
{
    __shared__ float red[256];
    const int row = blockIdx.x;               // (b*H + h)*S + i
    const int i = row & (S_ - 1);
    const int b = row / (H_ * S_);
    const float* qkr = qk + (long)row * S_;
    const float* qer = qe + (long)row * S_;
    unsigned short* ar = attn + (long)row * S_;
    const int* xb = x + b * S_;
    const int tid = threadIdx.x;

    float v[4];
    float mx = -3.0e38f;
    #pragma unroll
    for (int t = 0; t < 4; ++t) {
        const int j = tid + (t << 8);
        float lg = qkr[j];
        if (j <= i) lg += qer[i - j];
        lg *= scale;
        if (j > i || xb[j] == 0) lg -= 1e9f;
        v[t] = lg;
        mx = fmaxf(mx, lg);
    }
    const float m = blockReduceMax(mx, red);
    float s = 0.f;
    #pragma unroll
    for (int t = 0; t < 4; ++t) { v[t] = __expf(v[t] - m); s += v[t]; }
    const float tot = blockReduceSum(s, red);
    const float inv = 1.0f / tot;
    #pragma unroll
    for (int t = 0; t < 4; ++t) ar[tid + (t << 8)] = f2bf(v[t] * inv);
}

// ---------------------------------------------------------------------------
// LayerNorm over D per row: f32 in -> bf16 out
// ---------------------------------------------------------------------------
__global__ __launch_bounds__(256) void layernorm_k(
    const float* __restrict__ in, const float* __restrict__ g,
    const float* __restrict__ be, unsigned short* __restrict__ out)
{
    __shared__ float red[256];
    const int row = blockIdx.x;
    const int tid = threadIdx.x;
    const float* r = in + (long)row * D_;
    float xv[4]; float s = 0.f;
    #pragma unroll
    for (int t = 0; t < 4; ++t) { xv[t] = r[tid + (t << 8)]; s += xv[t]; }
    const float mu = blockReduceSum(s, red) * (1.0f / (float)D_);
    float q = 0.f;
    #pragma unroll
    for (int t = 0; t < 4; ++t) { const float d = xv[t] - mu; q += d * d; }
    const float var = blockReduceSum(q, red) * (1.0f / (float)D_);
    const float rs = rsqrtf(var + 1e-6f);
    #pragma unroll
    for (int t = 0; t < 4; ++t) {
        const int d = tid + (t << 8);
        out[(long)row * D_ + d] = f2bf((xv[t] - mu) * rs * g[d] + be[d]);
    }
}

// ---------------------------------------------------------------------------
// Host-side launcher: 128x128 tiles where possible, 64x64 otherwise
// ---------------------------------------------------------------------------
static inline void launch_gemm(hipStream_t st,
    const unsigned short* A, long lda, long sA1, long sA2,
    const unsigned short* Bm, long ldb, long sB1, long sB2,
    void* C, long ldc, long sC1, long sC2,
    const float* bias, int M, int N, int K,
    int batch, int binner, int relu, int outbf, int transC, int triskip)
{
    if ((M & 127) == 0 && (N & 127) == 0) {
        dim3 grid(N / 128, M / 128, batch);
        gemm_bf16_wmma<128, 128, 32, 64><<<grid, 256, 0, st>>>(
            A, lda, sA1, sA2, Bm, ldb, sB1, sB2, C, ldc, sC1, sC2,
            bias, M, N, K, binner, relu, outbf, transC, triskip);
    } else {
        dim3 grid(N / 64, M / 64, batch);
        gemm_bf16_wmma<64, 64, 32, 32><<<grid, 128, 0, st>>>(
            A, lda, sA1, sA2, Bm, ldb, sB1, sB2, C, ldc, sC1, sC2,
            bias, M, N, K, binner, relu, outbf, transC, triskip);
    }
}

extern "C" void kernel_launch(void* const* d_in, const int* in_sizes, int n_in,
                              void* d_out, int out_size, void* d_ws, size_t ws_size,
                              hipStream_t stream) {
    const int*   x   = (const int*)  d_in[0];
    const float* emb = (const float*)d_in[1];
    const float* Wq  = (const float*)d_in[2];
    const float* bq  = (const float*)d_in[3];
    const float* Wk  = (const float*)d_in[4];
    const float* bk  = (const float*)d_in[5];
    const float* Wv  = (const float*)d_in[6];
    const float* bv  = (const float*)d_in[7];
    const float* Wo  = (const float*)d_in[8];
    const float* bo  = (const float*)d_in[9];
    const float* W1  = (const float*)d_in[10];
    const float* b1  = (const float*)d_in[11];
    const float* W2  = (const float*)d_in[12];
    const float* b2  = (const float*)d_in[13];
    const float* g1  = (const float*)d_in[14];
    const float* be1 = (const float*)d_in[15];
    const float* g2  = (const float*)d_in[16];
    const float* be2 = (const float*)d_in[17];
    const float* E   = (const float*)d_in[18];
    const float* Wf  = (const float*)d_in[19];
    const float* bf  = (const float*)d_in[20];
    float* out = (float*)d_out;

    const long MBS = (long)B_ * S_;          // 2048
    const long DD  = (long)D_ * D_;
    const long DH  = (long)D_ * (D_ / 2);
    const long SS  = (long)S_ * S_;
    const long HSS = (long)H_ * SS;
    const long SD  = (long)S_ * D_;
    const long DS  = (long)D_ * S_;

    // ---- workspace carve-up (256B aligned chunks) ----
    char* wptr = (char*)d_ws;
    auto alloc = [&](size_t bytes) -> void* {
        void* p = (void*)wptr;
        wptr += (bytes + 255) & ~(size_t)255;
        return p;
    };
    typedef unsigned short u16;
    u16* hbuf = (u16*)alloc(MBS * D_ * 2);
    u16* qb   = (u16*)alloc(MBS * D_ * 2);
    u16* kb   = (u16*)alloc(MBS * D_ * 2);
    u16* vtb  = (u16*)alloc((long)B_ * DS * 2);      // V transposed [b][d][s]
    u16* cb   = (u16*)alloc(MBS * D_ * 2);
    u16* bb   = (u16*)alloc(MBS * D_ * 2);
    u16* f1b  = (u16*)alloc(MBS * (D_ / 2) * 2);
    u16* atb  = (u16*)alloc((long)B_ * HSS * 2);     // attention probs bf16
    u16* wqT  = (u16*)alloc((long)L_ * DD * 2);
    u16* wkT  = (u16*)alloc((long)L_ * DD * 2);
    u16* wvT  = (u16*)alloc((long)L_ * DD * 2);
    u16* woT  = (u16*)alloc((long)L_ * DD * 2);
    u16* w1T  = (u16*)alloc((long)L_ * DH * 2);
    u16* w2T  = (u16*)alloc((long)L_ * DH * 2);
    u16* wfT  = (u16*)alloc((long)D_ * V_ * 2);
    u16* erev = (u16*)alloc((long)L_ * S_ * HD_ * 2);
    float* ab  = (float*)alloc(MBS * D_ * 4);        // pre-LN activations f32
    float* qkb = (float*)alloc((long)B_ * HSS * 4);
    float* qeb = (float*)alloc((long)B_ * HSS * 4);

    // ---- one-time weight transpose+convert (bf16 [NxK]) ----
    transpose_cvt_k<<<dim3(32, 32, L_), 256, 0, stream>>>(Wq, wqT, D_, D_);
    transpose_cvt_k<<<dim3(32, 32, L_), 256, 0, stream>>>(Wk, wkT, D_, D_);
    transpose_cvt_k<<<dim3(32, 32, L_), 256, 0, stream>>>(Wv, wvT, D_, D_);
    transpose_cvt_k<<<dim3(32, 32, L_), 256, 0, stream>>>(Wo, woT, D_, D_);
    transpose_cvt_k<<<dim3(16, 32, L_), 256, 0, stream>>>(W1, w1T, D_, D_ / 2);
    transpose_cvt_k<<<dim3(32, 16, L_), 256, 0, stream>>>(W2, w2T, D_ / 2, D_);
    transpose_cvt_k<<<dim3(8, 32, 1),  256, 0, stream>>>(Wf, wfT, D_, V_);
    erev_k<<<(L_ * S_ * HD_) / 256, 256, 0, stream>>>(E, erev);

    // ---- embedding + positional encoding ----
    embed_k<<<(int)MBS, 256, 0, stream>>>(x, emb, hbuf);

    for (int l = 0; l < L_; ++l) {
        // Q,K projections: [2048x1024] @ W^T -> bf16
        launch_gemm(stream, hbuf, D_, 0, 0, wqT + (long)l * DD, D_, 0, 0,
                    qb, D_, 0, 0, bq + (long)l * D_,
                    (int)MBS, D_, D_, 1, 1, 0, 1, 0, 0);
        launch_gemm(stream, hbuf, D_, 0, 0, wkT + (long)l * DD, D_, 0, 0,
                    kb, D_, 0, 0, bk + (long)l * D_,
                    (int)MBS, D_, D_, 1, 1, 0, 1, 0, 0);
        // V projection, stored transposed per batch: vtb[b][d][s]
        launch_gemm(stream, hbuf, D_, 0, SD, wvT + (long)l * DD, D_, 0, 0,
                    vtb, S_, 0, DS, bv + (long)l * D_,
                    S_, D_, D_, B_, 1, 0, 1, 1, 0);

        // QK^T per (b,h): [Sx64] @ [Sx64]^T -> f32 (lower-triangular blocks)
        launch_gemm(stream, qb, D_, HD_, SD, kb, D_, HD_, SD,
                    qkb, S_, SS, HSS, nullptr, S_, S_, HD_,
                    B_ * H_, H_, 0, 0, 0, 1);

        // QE' per (b,h): [Sx64] @ erev[l] ([S x 64], N x K) -> f32
        launch_gemm(stream, qb, D_, HD_, SD, erev + (long)l * S_ * HD_, HD_, 0, 0,
                    qeb, S_, SS, HSS, nullptr, S_, S_, HD_,
                    B_ * H_, H_, 0, 0, 0, 1);

        // Fused skew-gather + mask + softmax -> attn bf16
        softmax_rel_k<<<B_ * H_ * S_, 256, 0, stream>>>(qkb, qeb, x, atb, 0.125f);

        // attn @ V per (b,h): [SxS] @ vtb[b][h*64..][S]^T -> context bf16
        launch_gemm(stream, atb, S_, SS, HSS, vtb, S_, (long)HD_ * S_, DS,
                    cb, D_, HD_, SD, nullptr, S_, HD_, S_,
                    B_ * H_, H_, 0, 1, 0, 0);

        // Output projection -> f32 (pre-LN)
        launch_gemm(stream, cb, D_, 0, 0, woT + (long)l * DD, D_, 0, 0,
                    ab, D_, 0, 0, bo + (long)l * D_,
                    (int)MBS, D_, D_, 1, 1, 0, 0, 0, 0);

        // LayerNorm 1 -> bf16
        layernorm_k<<<(int)MBS, 256, 0, stream>>>(ab, g1 + (long)l * D_,
                                                  be1 + (long)l * D_, bb);

        // FFN: relu(bb @ W1^T + b1) -> bf16 ; @ W2^T + b2 -> f32
        launch_gemm(stream, bb, D_, 0, 0, w1T + (long)l * DH, D_, 0, 0,
                    f1b, D_ / 2, 0, 0, b1 + (long)l * (D_ / 2),
                    (int)MBS, D_ / 2, D_, 1, 1, 1, 1, 0, 0);
        launch_gemm(stream, f1b, D_ / 2, 0, 0, w2T + (long)l * DH, D_ / 2, 0, 0,
                    ab, D_, 0, 0, b2 + (long)l * D_,
                    (int)MBS, D_, D_ / 2, 1, 1, 0, 0, 0, 0);

        // LayerNorm 2 -> next h (bf16)
        layernorm_k<<<(int)MBS, 256, 0, stream>>>(ab, g2 + (long)l * D_,
                                                  be2 + (long)l * D_, hbuf);
    }

    // Final projection to vocab: [2048x1024] @ Wf^T -> f32 out
    launch_gemm(stream, hbuf, D_, 0, 0, wfT, D_, 0, 0,
                out, V_, 0, 0, bf, (int)MBS, V_, D_, 1, 1, 0, 0, 0, 0);
}
